// SAPM_49349174231565
// MI455X (gfx1250) — compile-verified, matrix operand order checked by reading, not run on hardware
//
#include <hip/hip_runtime.h>
#include <hip/hip_fp16.h>

typedef __attribute__((ext_vector_type(16))) _Float16 v16h;
typedef __attribute__((ext_vector_type(8)))  _Float16 v8h;
typedef __attribute__((ext_vector_type(8)))  float    v8f;

// ---------------------------------------------------------------- helpers ---

__device__ __forceinline__ float fq_dev(float x, float a, float qn, float qp) {
    float r = rintf(x / a);            // round-half-even, matches jnp.round
    r = fminf(fmaxf(r, qn), qp);
    return r * a;
}

__device__ __forceinline__ v16h cat16(v8h lo, v8h hi) {
    return __builtin_shufflevector(lo, hi, 0,1,2,3,4,5,6,7,8,9,10,11,12,13,14,15);
}

#define WMMA_F16(A,B,C) \
    __builtin_amdgcn_wmma_f32_16x16x32_f16(false,(A),false,(B),(short)0,(C),false,false)

// ------------------------------------------------------- weight fake-quant ---

__global__ void fq_to_half_kernel(const float* __restrict__ w, const float* __restrict__ alpha,
                                  _Float16* __restrict__ out, int n) {
    int i = blockIdx.x * 256 + threadIdx.x;
    if (i < n) out[i] = (_Float16)fq_dev(w[i], alpha[0], -8.f, 7.f);
}

__global__ void fq_to_float_kernel(const float* __restrict__ w, const float* __restrict__ alpha,
                                   float* __restrict__ out, int n) {
    int i = blockIdx.x * 256 + threadIdx.x;
    if (i < n) out[i] = fq_dev(w[i], alpha[0], -8.f, 7.f);
}

// ------------------------------------------------------------ depthwise conv -
// Input: [B,256, ISTR] (ISTR=49 for raw F fp32, 64 for padded f16 activations)
// Output: padded half [B,256,64]; columns 49..63 are written as zero so the
// downstream WMMA GEMM can stage with unguarded 16-byte vector loads.

template<int KS, int PAD, int ISTR, typename TIN>
__global__ void dwconv_kernel(const TIN* __restrict__ x, const float* __restrict__ wq,
                              const float* __restrict__ bias, _Float16* __restrict__ out) {
    int idx = blockIdx.x * 256 + threadIdx.x;       // 2048*256*64 threads
    int s  = idx & 63;
    int bc = idx >> 6;
    int c  = bc & 255;
    float acc = 0.f;
    if (s < 49) {
        int oy = s / 7, ox = s % 7;
        const TIN*   xp = x  + (long)bc * ISTR;
        const float* wp = wq + c * KS * KS;
        acc = bias ? bias[c] : 0.f;
#pragma unroll
        for (int ky = 0; ky < KS; ++ky) {
            int iy = oy + ky - PAD;
            if (iy < 0 || iy >= 7) continue;
#pragma unroll
            for (int kx = 0; kx < KS; ++kx) {
                int ix = ox + kx - PAD;
                if (ix < 0 || ix >= 7) continue;
                acc += wp[ky * KS + kx] * (float)xp[iy * 7 + ix];
            }
        }
    }
    out[idx] = (_Float16)acc;                       // zero in the pad columns
}

// ------------------------------------- pointwise 1x1 GEMM + GroupNorm + PReLU
// Per-batch GEMM: Y[256co x 49s] = Wh[256x256] * Xh[256x64pad]  (WMMA f16)
// 8 waves/block; wave owns a 32(co) x 64(s) strip = 4 complete GN groups
// (group g = 8 channels; within a wave, group index = i*2 + lane_half), so
// GN+PReLU is fused in-register with a 16-lane shfl reduction. Output is the
// padded half layout consumed by the next depthwise conv.

__global__ void __launch_bounds__(256) pw_gemm_gn_kernel(const _Float16* __restrict__ Wh,
                                                         const _Float16* __restrict__ Xh,
                                                         const float* __restrict__ gamma,
                                                         const float* __restrict__ beta,
                                                         const float* __restrict__ pr,
                                                         _Float16* __restrict__ Out) {
    __shared__ __align__(16) _Float16 XT[64 * 32];  // [s][ci_local] transposed K-slice
    const int  tid  = threadIdx.x;
    const int  wave = tid >> 5;
    const int  lane = tid & 31;
    const int  hs   = lane >> 4;
    const int  ln   = lane & 15;
    const long xb   = (long)blockIdx.x * 16384;     // 256*64

    v8f acc[2][4];
#pragma unroll
    for (int i = 0; i < 2; ++i)
#pragma unroll
        for (int j = 0; j < 4; ++j)
#pragma unroll
            for (int r = 0; r < 8; ++r) acc[i][j][r] = 0.f;

    const int st_ci = tid >> 3;                     // staging: thread -> (ci, 8 s vals)
    const int st_s8 = (tid & 7) * 8;

    for (int kt = 0; kt < 8; ++kt) {
        const int k0 = kt * 32;
        __syncthreads();
        {   // one aligned 16B global load per thread, 8 scatter stores (transpose)
            v8h v = *(const v8h*)(Xh + xb + (long)(k0 + st_ci) * 64 + st_s8);
#pragma unroll
            for (int q = 0; q < 8; ++q) XT[(st_s8 + q) * 32 + st_ci] = v[q];
        }
        __syncthreads();

        v16h af[2], bf[4];
#pragma unroll
        for (int i = 0; i < 2; ++i) {               // A frag: W row m, K-contiguous
            const _Float16* p = Wh + ((wave * 2 + i) * 16 + ln) * 256 + k0 + hs * 8;
            af[i] = cat16(*(const v8h*)p, *(const v8h*)(p + 16));
        }
#pragma unroll
        for (int j = 0; j < 4; ++j) {               // B frag: col s, K-contiguous in XT
            const _Float16* p = XT + (j * 16 + ln) * 32 + hs * 8;
            bf[j] = cat16(*(const v8h*)p, *(const v8h*)(p + 16));
        }
#pragma unroll
        for (int i = 0; i < 2; ++i)
#pragma unroll
            for (int j = 0; j < 4; ++j)
                acc[i][j] = WMMA_F16(af[i], bf[j], acc[i][j]);
    }

    // ---- fused GroupNorm (+PReLU): stats over 8ch x 49s = 392 elements ----
    const float a = pr[0];
#pragma unroll
    for (int i = 0; i < 2; ++i) {
        float s1 = 0.f, s2 = 0.f;
#pragma unroll
        for (int j = 0; j < 4; ++j) {
            int s = j * 16 + ln;
            if (s < 49) {
#pragma unroll
                for (int r = 0; r < 8; ++r) { float v = acc[i][j][r]; s1 += v; s2 += v * v; }
            }
        }
#pragma unroll
        for (int off = 8; off; off >>= 1) {         // reduce across the 16-lane half
            s1 += __shfl_xor(s1, off, 32);
            s2 += __shfl_xor(s2, off, 32);
        }
        float m   = s1 / 392.f;
        float var = s2 / 392.f - m * m;
        float inv = rsqrtf(var + 1e-5f);
        float gm[8], bt[8];
#pragma unroll
        for (int r = 0; r < 8; ++r) {
            int co = wave * 32 + i * 16 + hs * 8 + r;
            gm[r] = gamma[co]; bt[r] = beta[co];
        }
#pragma unroll
        for (int j = 0; j < 4; ++j) {
            int s = j * 16 + ln;
#pragma unroll
            for (int r = 0; r < 8; ++r) {
                int   co = wave * 32 + i * 16 + hs * 8 + r;
                float v  = (acc[i][j][r] - m) * inv * gm[r] + bt[r];
                v = (v >= 0.f) ? v : a * v;
                Out[xb + co * 64 + s] = (s < 49) ? (_Float16)v : (_Float16)0.f;
            }
        }
    }
}

// ------------------------------------- c3 pointwise + softmax(tau) + fq(A) ---
// t6: half [B,256,64pad]; A[b,s] = sum_ci t6*w + b, softmax over s, per-s fq.

__global__ void __launch_bounds__(64) softmaxA_kernel(const _Float16* __restrict__ t6,
                                                      const float* __restrict__ pw,
                                                      const float* __restrict__ pwb,
                                                      const float* __restrict__ Aa,
                                                      float* __restrict__ Aq) {
    __shared__ float red[64];
    const int b = blockIdx.x, t = threadIdx.x;
    float acc = -1e30f;
    if (t < 49) {
        const _Float16* x = t6 + (long)b * 16384 + t;
        float v = 0.f;
        for (int ci = 0; ci < 256; ++ci) v += (float)x[ci * 64] * pw[ci];
        acc = (v + pwb[0]) * 1.2f;                  // * tau before softmax
    }
    red[t] = acc; __syncthreads();
#pragma unroll
    for (int o = 32; o; o >>= 1) { if (t < o) red[t] = fmaxf(red[t], red[t + o]); __syncthreads(); }
    float mx = red[0]; __syncthreads();
    float e = (t < 49) ? __expf(acc - mx) : 0.f;
    red[t] = e; __syncthreads();
#pragma unroll
    for (int o = 32; o; o >>= 1) { if (t < o) red[t] += red[t + o]; __syncthreads(); }
    float sum = red[0];
    if (t < 49) Aq[b * 49 + t] = fq_dev(e / sum, Aa[t], -8.f, 7.f);
}

// ------------------------------------------------ WP pooling: F_P = Aq @ Fq --

__global__ void __launch_bounds__(256) pool_kernel(const float* __restrict__ F,
                                                   const float* __restrict__ Aq,
                                                   const float* __restrict__ Fa,
                                                   float* __restrict__ FP,
                                                   _Float16* __restrict__ FPh) {
    __shared__ float As[49];
    const int b = blockIdx.x, c = threadIdx.x;
    if (c < 49) As[c] = Aq[b * 49 + c];
    __syncthreads();
    const float* x = F + (long)b * 12544 + c * 49;
    const float fa = Fa[0];
    float acc = 0.f;
#pragma unroll
    for (int s = 0; s < 49; ++s) acc += As[s] * fq_dev(x[s], fa, 0.f, 15.f);
    FP[b * 256 + c]  = acc;
    FPh[b * 256 + c] = (_Float16)acc;
}

// ---------------------------------------------------------- FC WMMA GEMMs ----
// Y[2048,256] = A[2048,256] * W^T, both f16 row-major k-contiguous.
// Grid (32,2) x 256 thr; wave = 32x32 tile; block = 64(M) x 128(N).

__global__ void __launch_bounds__(256) fc1_kernel(const _Float16* __restrict__ Ah,
                                                  const _Float16* __restrict__ Bh,
                                                  const float* __restrict__ bias,
                                                  const float* __restrict__ pr,
                                                  _Float16* __restrict__ H1) {
    const int tid = threadIdx.x, wave = tid >> 5, lane = tid & 31;
    const int hs = lane >> 4, ln = lane & 15;
    const int mt0 = blockIdx.x * 4 + (wave >> 2) * 2;
    const int nt0 = blockIdx.y * 8 + (wave & 3) * 2;
    v8f acc[2][2];
#pragma unroll
    for (int i = 0; i < 2; ++i)
#pragma unroll
        for (int j = 0; j < 2; ++j)
#pragma unroll
            for (int r = 0; r < 8; ++r) acc[i][j][r] = 0.f;

    for (int kt = 0; kt < 8; ++kt) {
        const int k0 = kt * 32;
        v16h af[2], bf[2];
#pragma unroll
        for (int i = 0; i < 2; ++i) {
            const _Float16* p = Ah + ((mt0 + i) * 16 + ln) * 256 + k0 + hs * 8;
            af[i] = cat16(*(const v8h*)p, *(const v8h*)(p + 16));
        }
#pragma unroll
        for (int j = 0; j < 2; ++j) {
            const _Float16* p = Bh + ((nt0 + j) * 16 + ln) * 256 + k0 + hs * 8;
            bf[j] = cat16(*(const v8h*)p, *(const v8h*)(p + 16));
        }
#pragma unroll
        for (int i = 0; i < 2; ++i)
#pragma unroll
            for (int j = 0; j < 2; ++j)
                acc[i][j] = WMMA_F16(af[i], bf[j], acc[i][j]);
    }
    const float a = pr[0];
#pragma unroll
    for (int i = 0; i < 2; ++i)
#pragma unroll
        for (int j = 0; j < 2; ++j)
#pragma unroll
            for (int r = 0; r < 8; ++r) {
                int m = (mt0 + i) * 16 + hs * 8 + r;
                int n = (nt0 + j) * 16 + ln;
                float v = acc[i][j][r] + bias[n];
                H1[m * 256 + n] = (_Float16)((v >= 0.f) ? v : a * v);
            }
}

__global__ void __launch_bounds__(256) fc2_kernel(const _Float16* __restrict__ Ah,
                                                  const _Float16* __restrict__ Bh,
                                                  const float* __restrict__ bias,
                                                  const float* __restrict__ FP,
                                                  const float* __restrict__ FCa,
                                                  const float* __restrict__ FPa,
                                                  float* __restrict__ Out) {
    const int tid = threadIdx.x, wave = tid >> 5, lane = tid & 31;
    const int hs = lane >> 4, ln = lane & 15;
    const int mt0 = blockIdx.x * 4 + (wave >> 2) * 2;
    const int nt0 = blockIdx.y * 8 + (wave & 3) * 2;
    v8f acc[2][2];
#pragma unroll
    for (int i = 0; i < 2; ++i)
#pragma unroll
        for (int j = 0; j < 2; ++j)
#pragma unroll
            for (int r = 0; r < 8; ++r) acc[i][j][r] = 0.f;

    for (int kt = 0; kt < 8; ++kt) {
        const int k0 = kt * 32;
        v16h af[2], bf[2];
#pragma unroll
        for (int i = 0; i < 2; ++i) {
            const _Float16* p = Ah + ((mt0 + i) * 16 + ln) * 256 + k0 + hs * 8;
            af[i] = cat16(*(const v8h*)p, *(const v8h*)(p + 16));
        }
#pragma unroll
        for (int j = 0; j < 2; ++j) {
            const _Float16* p = Bh + ((nt0 + j) * 16 + ln) * 256 + k0 + hs * 8;
            bf[j] = cat16(*(const v8h*)p, *(const v8h*)(p + 16));
        }
#pragma unroll
        for (int i = 0; i < 2; ++i)
#pragma unroll
            for (int j = 0; j < 2; ++j)
                acc[i][j] = WMMA_F16(af[i], bf[j], acc[i][j]);
    }
    const float fca = FCa[0], fpa = FPa[0];
#pragma unroll
    for (int i = 0; i < 2; ++i)
#pragma unroll
        for (int j = 0; j < 2; ++j)
#pragma unroll
            for (int r = 0; r < 8; ++r) {
                int m = (mt0 + i) * 16 + hs * 8 + r;
                int n = (nt0 + j) * 16 + ln;
                float v  = acc[i][j][r] + bias[n];
                float sg = 1.f / (1.f + __expf(-v));
                float fc = fq_dev(sg, fca, 0.f, 15.f);
                float fp = fq_dev(FP[m * 256 + n], fpa, 0.f, 15.f);
                Out[m * 256 + n] = fc * fp + fp;
            }
}

// ------------------------------------------------------------------- launch --

extern "C" void kernel_launch(void* const* d_in, const int* in_sizes, int n_in,
                              void* d_out, int out_size, void* d_ws, size_t ws_size,
                              hipStream_t stream) {
    (void)in_sizes; (void)n_in; (void)out_size; (void)ws_size;
    const float* F       = (const float*)d_in[0];
    const float* c1_dw   = (const float*)d_in[1];
    const float* c1_pw   = (const float*)d_in[2];
    const float* c2_dw   = (const float*)d_in[3];
    const float* c2_pw   = (const float*)d_in[4];
    const float* c3_dw   = (const float*)d_in[5];
    const float* c3_dw_b = (const float*)d_in[6];
    const float* c3_pw   = (const float*)d_in[7];
    const float* c3_pw_b = (const float*)d_in[8];
    const float* c1_dw_a = (const float*)d_in[9];
    const float* c1_pw_a = (const float*)d_in[10];
    const float* c2_dw_a = (const float*)d_in[11];
    const float* c2_pw_a = (const float*)d_in[12];
    const float* c3_dw_a = (const float*)d_in[13];
    const float* c3_pw_a = (const float*)d_in[14];
    const float* gn1_g   = (const float*)d_in[15];
    const float* gn1_b   = (const float*)d_in[16];
    const float* gn2_g   = (const float*)d_in[17];
    const float* gn2_b   = (const float*)d_in[18];
    const float* pr1     = (const float*)d_in[19];
    const float* pr2     = (const float*)d_in[20];
    const float* pr3     = (const float*)d_in[21];
    const float* fc1_w   = (const float*)d_in[22];
    const float* fc1_b   = (const float*)d_in[23];
    const float* fc2_w   = (const float*)d_in[24];
    const float* fc2_b   = (const float*)d_in[25];
    const float* fc1_w_a = (const float*)d_in[26];
    const float* fc2_w_a = (const float*)d_in[27];
    const float* A_act_a = (const float*)d_in[28];
    const float* F_act_a = (const float*)d_in[29];
    const float* FC_act_a= (const float*)d_in[30];
    const float* FP_act_a= (const float*)d_in[31];

    char* ws = (char*)d_ws;
    float*    c1dwq = (float*)(ws + 0);          // 6400 f32
    float*    c2dwq = (float*)(ws + 25600);      // 2304 f32
    float*    c3dwq = (float*)(ws + 34816);      // 2304 f32
    float*    c3pwq = (float*)(ws + 44032);      // 256 f32
    _Float16* W1h   = (_Float16*)(ws + 45056);   // 65536 f16 each
    _Float16* W2h   = (_Float16*)(ws + 176128);
    _Float16* FC1h  = (_Float16*)(ws + 307200);
    _Float16* FC2h  = (_Float16*)(ws + 438272);
    float*    Aq    = (float*)(ws + 569344);     // 2048*49 f32
    float*    FP    = (float*)(ws + 970752);     // 2048*256 f32
    _Float16* FPh   = (_Float16*)(ws + 3067904); // 2048*256 f16
    _Float16* H1h   = (_Float16*)(ws + 4116480); // 2048*256 f16
    _Float16* bufA  = (_Float16*)(ws + 5165056); // 2048*256*64 f16 (67MB)
    _Float16* bufB  = (_Float16*)(ws + 72273920);// 2048*256*64 f16 (67MB)

    // --- fake-quantize weights once ---
    fq_to_half_kernel <<<256, 256, 0, stream>>>(c1_pw, c1_pw_a, W1h, 65536);
    fq_to_half_kernel <<<256, 256, 0, stream>>>(c2_pw, c2_pw_a, W2h, 65536);
    fq_to_half_kernel <<<256, 256, 0, stream>>>(fc1_w, fc1_w_a, FC1h, 65536);
    fq_to_half_kernel <<<256, 256, 0, stream>>>(fc2_w, fc2_w_a, FC2h, 65536);
    fq_to_float_kernel<<<25, 256, 0, stream>>>(c1_dw, c1_dw_a, c1dwq, 6400);
    fq_to_float_kernel<<<9, 256, 0, stream>>>(c2_dw, c2_dw_a, c2dwq, 2304);
    fq_to_float_kernel<<<9, 256, 0, stream>>>(c3_dw, c3_dw_a, c3dwq, 2304);
    fq_to_float_kernel<<<1, 256, 0, stream>>>(c3_pw, c3_pw_a, c3pwq, 256);

    // --- AMP branch: dw -> (pw GEMM + GN + PReLU fused) x2 -> dw -> softmax ---
    dwconv_kernel<5, 2, 49, float>   <<<131072, 256, 0, stream>>>(F, c1dwq, nullptr, bufA);
    pw_gemm_gn_kernel<<<2048, 256, 0, stream>>>(W1h, bufA, gn1_g, gn1_b, pr1, bufB);
    dwconv_kernel<3, 1, 64, _Float16><<<131072, 256, 0, stream>>>(bufB, c2dwq, nullptr, bufA);
    pw_gemm_gn_kernel<<<2048, 256, 0, stream>>>(W2h, bufA, gn2_g, gn2_b, pr2, bufB);
    dwconv_kernel<3, 1, 64, _Float16><<<131072, 256, 0, stream>>>(bufB, c3dwq, c3_dw_b, bufA);
    softmaxA_kernel<<<2048, 64, 0, stream>>>(bufA, c3pwq, c3_pw_b, A_act_a, Aq);

    // --- WP pooling + CR + SAPM combine ---
    pool_kernel<<<2048, 256, 0, stream>>>(F, Aq, F_act_a, FP, FPh);
    fc1_kernel<<<dim3(32, 2), 256, 0, stream>>>(FPh, FC1h, fc1_b, pr3, H1h);
    fc2_kernel<<<dim3(32, 2), 256, 0, stream>>>(H1h, FC2h, fc2_b, FP, FC_act_a, FP_act_a,
                                                (float*)d_out);
}
